// LateInteraction_5488968204794
// MI455X (gfx1250) — matrix operand order repaired and checked by reading, not compiled, not run on hardware
//
#include <hip/hip_runtime.h>
#include <hip/hip_bf16.h>
#include <math.h>

// ---------------------------------------------------------------------------
// LateInteraction: out[i,j] = sum_s qmask[i,s] * softmax_t(mask? qn·kn*e^{-a|s-t|} : -inf) · (qn·kn)
// B=8, KB=16, SEQ=512, HID=768.  Compute-bound (51.5 GFLOP GEMM), all data L2-resident.
// Path: bf16 WMMA (v_wmma_f32_16x16x32_bf16) + flash-style online softmax.
// ---------------------------------------------------------------------------

#define B_   8
#define KB_  16
#define SEQ_ 512
#define HID_ 768

typedef __attribute__((ext_vector_type(16))) __bf16 v16bf;
typedef __attribute__((ext_vector_type(8)))  __bf16 v8bf;
typedef __attribute__((ext_vector_type(8)))  float  v8f;

// ---------------- kernel 0: zero output -----------------------------------
__global__ void li_zero(float* __restrict__ out) {
    if (threadIdx.x < B_ * KB_) out[threadIdx.x] = 0.0f;
}

// ---------------- kernel 1: fused L2-normalize + fp32->bf16 ----------------
// One wave per row of 768. Rows 0..4095 = Q, rows 4096..12287 = K.
__global__ __launch_bounds__(256) void li_norm(const float* __restrict__ q,
                                               const float* __restrict__ k,
                                               __bf16* __restrict__ qn,
                                               __bf16* __restrict__ kn) {
    const int lane = threadIdx.x & 31;
    const int row  = blockIdx.x * 8 + (threadIdx.x >> 5);

    const float* src;
    __bf16* dst;
    if (row < B_ * SEQ_) {
        src = q  + (size_t)row * HID_;
        dst = qn + (size_t)row * HID_;
    } else {
        const int r2 = row - B_ * SEQ_;
        src = k  + (size_t)r2 * HID_;
        dst = kn + (size_t)r2 * HID_;
    }

    float vals[24];
    float ss = 0.0f;
#pragma unroll
    for (int c = 0; c < 24; ++c) {
        float v = src[lane + 32 * c];
        vals[c] = v;
        ss = fmaf(v, v, ss);
    }
#pragma unroll
    for (int m = 1; m < 32; m <<= 1) ss += __shfl_xor(ss, m, 32);

    const float inv = 1.0f / fmaxf(sqrtf(ss), 1e-12f);
#pragma unroll
    for (int c = 0; c < 24; ++c) dst[lane + 32 * c] = (__bf16)(vals[c] * inv);
}

// ---------------- fragment loaders (ISA 16-bit A/B VGPR layouts) -----------
// A 16x32 bf16: lane<16 holds M=lane, K=[0..7]+[16..23]; lane>=16 K=[8..15]+[24..31]
__device__ __forceinline__ v16bf load_frag_a(const __bf16* p) {
    union { v16bf v; v8bf h[2]; } u;
    u.h[0] = *(const v8bf*)(p);
    u.h[1] = *(const v8bf*)(p + 16);
    return u.v;
}
// B 32x16 bf16: lane<16 holds N=lane, K=[0..7]+[16..23]; lane>=16 K=[8..15]+[24..31]
__device__ __forceinline__ v16bf load_frag_b(const __bf16* p) {
    union { v16bf v; v8bf h[2]; } u;
    u.h[0] = *(const v8bf*)(p);
    u.h[1] = *(const v8bf*)(p + 16);
    return u.v;
}

// ---------------- online softmax update for one 16x16 C tile ---------------
// C layout: lane holds (M = r + 8*(lane>=16), N = lane&15) in c[r], r=0..7.
__device__ __forceinline__ void online_update(v8f c, int s0, int tcol, int hi,
                                              float alpha, const float* qmf,
                                              bool kvalid,
                                              float* m_s, float* l_s, float* acc_s) {
#pragma unroll
    for (int r = 0; r < 8; ++r) {
        const float sim  = c[r];
        const float srow = (float)(s0 + r + 8 * hi);
        const float w    = __expf(-alpha * fabsf(srow - (float)tcol));
        const float logit = (qmf[r] != 0.0f && kvalid) ? sim * w : -INFINITY;

        float tm = logit;
#pragma unroll
        for (int m = 1; m < 16; m <<= 1) tm = fmaxf(tm, __shfl_xor(tm, m, 32));
        const float mnew = fmaxf(m_s[r], tm);

        if (mnew != -INFINITY) {                // uniform within each 16-lane half
            const float scale = __expf(m_s[r] - mnew);   // exp(-inf)=0 handles init
            const float e     = __expf(logit - mnew);    // exp(-inf)=0 handles mask
            float se = e, ses = e * sim;
#pragma unroll
            for (int m = 1; m < 16; m <<= 1) {
                se  += __shfl_xor(se,  m, 32);
                ses += __shfl_xor(ses, m, 32);
            }
            l_s[r]   = l_s[r]   * scale + se;
            acc_s[r] = acc_s[r] * scale + ses;
            m_s[r]   = mnew;
        }
    }
}

// ---------------- kernel 2: WMMA GEMM + online softmax + reduce ------------
// grid = B*KB*4 blocks, 256 threads (8 wave32). Wave owns 16 s-rows.
__global__ __launch_bounds__(256) void li_main(const __bf16* __restrict__ qn,
                                               const __bf16* __restrict__ kn,
                                               const int* __restrict__ qmask,
                                               const int* __restrict__ kmask,
                                               const float* __restrict__ alpha_raw,
                                               float* __restrict__ out) {
    const int lane = threadIdx.x & 31;
    const int wave = threadIdx.x >> 5;
    const int hi   = lane >> 4;          // 0 or 1 (wave half)
    const int ln   = lane & 15;

    const int ij   = blockIdx.x >> 2;    // 0..127
    const int i    = ij >> 4;
    const int j    = ij & 15;
    const int sblk = blockIdx.x & 3;
    const int s0   = (sblk * 8 + wave) * 16;   // this wave's 16-row s tile

    const float a0    = alpha_raw[0];
    const float alpha = a0 >= 0.0f ? a0 : 0.01f * a0;   // leaky_relu

    // per-row query mask (as float, multiplied in at the end like the ref)
    float qmf[8];
#pragma unroll
    for (int r = 0; r < 8; ++r)
        qmf[r] = (float)qmask[i * SEQ_ + s0 + r + 8 * hi];

    float m_s[8], l_s[8], acc_s[8];
#pragma unroll
    for (int r = 0; r < 8; ++r) { m_s[r] = -INFINITY; l_s[r] = 0.0f; acc_s[r] = 0.0f; }

    const __bf16* qbase = qn + (size_t)(i * SEQ_) * HID_;
    const __bf16* kbase = kn + (size_t)(j * SEQ_) * HID_;

    // A fragment base for this wave's rows (layout offset folded in)
    const __bf16* aptr = qbase + (size_t)(s0 + ln) * HID_ + hi * 8;

    // 32 t-tiles processed in pairs: A fragment feeds two WMMAs
    for (int tp = 0; tp < 16; ++tp) {
        const int t0 = tp * 32;
        const int tcol0 = t0 + ln;
        const int tcol1 = t0 + 16 + ln;
        const __bf16* bptr0 = kbase + (size_t)tcol0 * HID_ + hi * 8;
        const __bf16* bptr1 = kbase + (size_t)tcol1 * HID_ + hi * 8;

        v8f c0 = {}; v8f c1 = {};
#pragma unroll
        for (int kc = 0; kc < 24; ++kc) {
            const int ka = kc * 32;
            v16bf af = load_frag_a(aptr + ka);
            v16bf b0 = load_frag_b(bptr0 + ka);
            v16bf b1 = load_frag_b(bptr1 + ka);
            c0 = __builtin_amdgcn_wmma_f32_16x16x32_bf16(false, af, false, b0,
                                                         (short)0, c0, false, false);
            c1 = __builtin_amdgcn_wmma_f32_16x16x32_bf16(false, af, false, b1,
                                                         (short)0, c1, false, false);
        }

        const bool kv0 = kmask[j * SEQ_ + tcol0] != 0;
        const bool kv1 = kmask[j * SEQ_ + tcol1] != 0;
        online_update(c0, s0, tcol0, hi, alpha, qmf, kv0, m_s, l_s, acc_s);
        online_update(c1, s0, tcol1, hi, alpha, qmf, kv1, m_s, l_s, acc_s);
    }

    // finalize: score = acc/l (0 if empty row), * qmask, reduce rows, atomic add
    float tot = 0.0f;
#pragma unroll
    for (int r = 0; r < 8; ++r) {
        const float sc = (l_s[r] > 0.0f) ? (acc_s[r] / l_s[r]) : 0.0f;
        tot += sc * qmf[r];
    }
    if (ln == 0) atomicAdd(&out[ij], tot);   // lane 0 (rows 0-7) and lane 16 (rows 8-15)
}

// ---------------------------------------------------------------------------
extern "C" void kernel_launch(void* const* d_in, const int* in_sizes, int n_in,
                              void* d_out, int out_size, void* d_ws, size_t ws_size,
                              hipStream_t stream) {
    const float* q      = (const float*)d_in[0];   // [8,512,768] f32
    const float* k      = (const float*)d_in[1];   // [16,512,768] f32
    const float* alpha  = (const float*)d_in[2];   // scalar
    const int*   qmask  = (const int*)d_in[3];     // [8,512]
    const int*   kmask  = (const int*)d_in[4];     // [16,512]
    float*       out    = (float*)d_out;           // [8,16]

    __bf16* qn = (__bf16*)d_ws;                                        // 6.3 MB
    __bf16* kn = (__bf16*)((char*)d_ws + (size_t)B_ * SEQ_ * HID_ * 2); // 12.6 MB

    li_zero<<<1, 128, 0, stream>>>(out);
    li_norm<<<(B_ + KB_) * SEQ_ / 8, 256, 0, stream>>>(q, k, qn, kn);
    li_main<<<B_ * KB_ * 4, 256, 0, stream>>>(qn, kn, qmask, kmask, alpha, out);
}
